// CausalBindingMechanism_90091234001462
// MI455X (gfx1250) — compile-verified, matrix operand order checked by reading, not run on hardware
//
#include <hip/hip_runtime.h>
#include <hip/hip_bf16.h>
#include <cstdint>

// ---------------------------------------------------------------------------
// CausalBindingMechanism forward for MI455X (gfx1250, wave32, WMMA + TDM)
// ---------------------------------------------------------------------------

#define D_MODEL 2048
#define NUM_HEADS 16
#define HEAD_DIM 128
#define BATCH 8
#define NUM_SLOTS 16
#define SEQ_LEN 512

typedef _Float16 v16h __attribute__((ext_vector_type(16)));
typedef float    v8f  __attribute__((ext_vector_type(8)));

__device__ __forceinline__ float gelu_exact(float x) {
    return 0.5f * x * (1.0f + erff(x * 0.70710678118654752440f));
}

// ---------------------------------------------------------------------------
// Weight prep: fp32 (K x N) -> f16 TRANSPOSED (N x K), tiled through LDS so
// both global streams are coalesced. Transposed storage makes every GEMM
// B-tile a run of contiguous K-halfs per output column -> TDM can DMA it
// linearly and the padded LDS layout matches the WMMA fragment reads.
// ---------------------------------------------------------------------------
__global__ __launch_bounds__(256)
void cvt_transpose_f16(const float* __restrict__ src, _Float16* __restrict__ dst,
                       int K, int N)   // src: K x N, dst: N x K
{
    __shared__ _Float16 tile[32][33];
    const int n0 = blockIdx.x * 32;
    const int k0 = blockIdx.y * 32;
    const int tx = threadIdx.x & 31;
    const int ty = threadIdx.x >> 5;    // 0..7
    for (int i = ty; i < 32; i += 8)
        tile[i][tx] = (_Float16)src[(long)(k0 + i) * N + (n0 + tx)];
    __syncthreads();
    for (int i = ty; i < 32; i += 8)
        dst[(long)(n0 + i) * K + (k0 + tx)] = tile[tx][i];
}

// ---------------------------------------------------------------------------
// TDM descriptor issue: DMA a (BK x BN) weight tile (rows = output columns,
// 32 contiguous f16 K-values each) from WT[N x K] into LDS with 1-DWORD pad
// every 16 DWORDs -> LDS row stride of 34 halfs.
// ---------------------------------------------------------------------------
#if __has_builtin(__builtin_amdgcn_tensor_load_to_lds)
#define HAS_TDM 1
typedef unsigned int u32x4 __attribute__((ext_vector_type(4)));
typedef int          i32x8 __attribute__((ext_vector_type(8)));
typedef int          i32x4 __attribute__((ext_vector_type(4)));

__device__ __forceinline__ void tdm_load_wtile(const _Float16* WT, _Float16* ldsdst,
                                               int N, int K, int n0, int k0, int ncols)
{
    unsigned long long ga = (unsigned long long)(uintptr_t)(WT + (long)n0 * K + k0);
    unsigned lds = (unsigned)(uintptr_t)ldsdst;   // generic addr low 32 bits = LDS offset

    u32x4 g0;
    g0[0] = 1u;                                          // count=1, user mode
    g0[1] = lds;                                         // lds_addr
    g0[2] = (unsigned)(ga & 0xffffffffu);                // global_addr[31:0]
    g0[3] = (unsigned)((ga >> 32) & 0x1ffffffu)          // global_addr[56:32]
          | (2u << 30);                                  // type = 2 (image)

    i32x8 g1;
    g1[0] = (int)((1u << 16)      // data_size = 2 bytes
                | (1u << 20)      // pad_enable
                | (3u << 22));    // pad_interval: 16 DWORDs (one 32-half row)
                                  // pad_amount: code 0 = 1 DWORD -> stride 34 halfs
    g1[1] = (int)(((unsigned)K & 0xffffu) << 16);                         // tensor_dim0[15:0]
    g1[2] = (int)((((unsigned)K >> 16) & 0xffffu)
                | (((unsigned)N & 0xffffu) << 16));                       // dim0 hi | dim1 lo
    g1[3] = (int)((((unsigned)N >> 16) & 0xffffu) | (32u << 16));         // dim1 hi | tile_dim0=32
    g1[4] = (int)(unsigned)ncols;                                         // tile_dim1 (=BN)
    g1[5] = (int)K;                                                       // tensor_dim0_stride lo
    g1[6] = 0;
    g1[7] = 0;

    i32x4 z4 = {0, 0, 0, 0};
    i32x8 z8 = {0, 0, 0, 0, 0, 0, 0, 0};
    __builtin_amdgcn_tensor_load_to_lds(g0, g1, z4, z4, z8, 0);
}
#else
#define HAS_TDM 0
#endif

// ---------------------------------------------------------------------------
// WMMA GEMM:  C[M,N] = act( A[M,K] @ W[K,N] + bias )
//   * W supplied TRANSPOSED as WT[N x K] f16
//   * A optionally split into two fp32 sources along K (virtual concat)
//   * optional GELU applied to A while converting to f16
//   * optional sigmoid on result; optional fused blend C = g*v + (1-g)*res
//
// Block: 256 threads = 8 waves (wave32). Tile 128x128, BK=32.
// Waves laid out 4(M) x 2(N); each wave: 32x64 strip =
//   2 A-frags + 4 B-frags -> 8 v_wmma_f32_16x16x32_f16 per K-step.
// B tile staged by the Tensor Data Mover (wave 0 issues, s_wait_tensorcnt).
// ---------------------------------------------------------------------------
#define BM 128
#define BN 128
#define BK 32
#define LDST 34   // padded row stride (halfs): aligned u32 reads, no conflicts

__global__ __launch_bounds__(256)
void gemm_wmma_f16(const float* __restrict__ A1, int lda1,
                   const float* __restrict__ A2, int lda2, int splitK,
                   const _Float16* __restrict__ WT,       // N x K
                   const float* __restrict__ bias,
                   float* __restrict__ C,
                   int M, int N, int K,
                   int a_gelu, int act_sigmoid,
                   const float* __restrict__ ep_gate, const float* __restrict__ ep_res)
{
    __shared__ __align__(16) _Float16 sA[BM * LDST];   // [row][k]
    __shared__ __align__(16) _Float16 sB[BN * LDST];   // [col][k]

    const int tid  = threadIdx.x;
    const int lane = tid & 31;
    const int wid  = tid >> 5;
    const int wm   = wid >> 1;          // 0..3 -> rows wm*32
    const int wn   = wid & 1;           // 0..1 -> cols wn*64
    const int m0   = blockIdx.y * BM;
    const int n0   = blockIdx.x * BN;

    const int r16   = lane & 15;
    const int hi    = lane >> 4;
    const int kbase = hi * 8;           // ISA 7.12.2 16-bit fragment K grouping

    v8f acc[2][4] = {};

    for (int k0 = 0; k0 < K; k0 += BK) {
#if HAS_TDM
        if (wid == 0)                   // one TDM issue per block (EXEC ignored)
            tdm_load_wtile(WT, sB, N, K, n0, k0, BN);
#endif
        // ---- stage A tile (BM x BK) fp32 -> f16, optional GELU, split-K concat
        #pragma unroll
        for (int e = 0; e < 16; ++e) {
            int idx  = tid * 16 + e;          // 0..4095
            int row  = idx >> 5;
            int col  = idx & 31;
            int grow = m0 + row;
            int gk   = k0 + col;
            float v = 0.0f;
            if (grow < M) {
                if (gk < splitK) v = A1[(long)grow * lda1 + gk];
                else             v = A2[(long)grow * lda2 + (gk - splitK)];
            }
            if (a_gelu) v = gelu_exact(v);
            sA[row * LDST + col] = (_Float16)v;
        }
#if !HAS_TDM
        // ---- fallback: manual B staging from transposed weights
        #pragma unroll
        for (int e = 0; e < 16; ++e) {
            int idx = tid * 16 + e;           // 0..4095
            int cc  = idx >> 5;               // 0..127
            int kk  = idx & 31;
            sB[cc * LDST + kk] = WT[(long)(n0 + cc) * K + (k0 + kk)];
        }
#endif
#if HAS_TDM
        if (wid == 0)
            __builtin_amdgcn_s_wait_tensorcnt(0);   // TDM tile landed in LDS
#endif
        __syncthreads();

        // ---- fragments (aligned u32 pair loads, ISA 16-bit A/B layouts)
        union Frag { unsigned int u[8]; v16h h; } fa[2], fb[4];
        #pragma unroll
        for (int j = 0; j < 8; ++j) {
            int kk = (j < 4) ? (kbase + 2 * j) : (kbase + 16 + 2 * (j - 4));
            fa[0].u[j] = *(const unsigned int*)&sA[(wm * 32 +      r16) * LDST + kk];
            fa[1].u[j] = *(const unsigned int*)&sA[(wm * 32 + 16 + r16) * LDST + kk];
            #pragma unroll
            for (int t = 0; t < 4; ++t)
                fb[t].u[j] = *(const unsigned int*)&sB[(wn * 64 + t * 16 + r16) * LDST + kk];
        }

        #pragma unroll
        for (int mi = 0; mi < 2; ++mi)
            #pragma unroll
            for (int t = 0; t < 4; ++t)
                acc[mi][t] = __builtin_amdgcn_wmma_f32_16x16x32_f16(
                    false, fa[mi].h, false, fb[t].h, (short)0, acc[mi][t], false, false);

        if (k0 + BK < K)   // warm GL2 for next A rows (global_prefetch_b8)
            __builtin_prefetch(&A1[(long)(m0 < M ? m0 : 0) * lda1 + (k0 + BK < splitK ? k0 + BK : 0)], 0, 1);

        __syncthreads();
    }

    // ---- epilogue: C/D layout (lane 0-15: N=lane; vgpr r -> M = r + 8*hi)
    #pragma unroll
    for (int mi = 0; mi < 2; ++mi) {
        #pragma unroll
        for (int t = 0; t < 4; ++t) {
            #pragma unroll
            for (int r = 0; r < 8; ++r) {
                int grow = m0 + wm * 32 + mi * 16 + hi * 8 + r;
                if (grow >= M) continue;
                int gcol = n0 + wn * 64 + t * 16 + r16;
                float v = acc[mi][t][r] + bias[gcol];
                if (act_sigmoid) v = 1.0f / (1.0f + expf(-v));
                long o = (long)grow * N + gcol;
                if (ep_gate) { float g = ep_gate[o]; v = g * v + (1.0f - g) * ep_res[o]; }
                C[o] = v;
            }
        }
    }
}

// ---------------------------------------------------------------------------
// LayerNorm over last dim D (=2048). One block per row. res may be null;
// res row index = row / res_div. Supports in-place (out == x).
// ---------------------------------------------------------------------------
__global__ __launch_bounds__(256)
void ln_kernel(const float* __restrict__ x, const float* __restrict__ res, int res_div,
               const float* __restrict__ g, const float* __restrict__ b,
               float* __restrict__ out, int D)
{
    __shared__ float s1[256];
    __shared__ float s2[256];
    const int row = blockIdx.x;
    const int tid = threadIdx.x;
    const float* xr = x + (long)row * D;
    const float* rr = res ? res + (long)(row / res_div) * D : nullptr;

    float a = 0.0f, q = 0.0f;
    for (int d = tid; d < D; d += 256) {
        float v = xr[d] + (rr ? rr[d] : 0.0f);
        a += v; q += v * v;
    }
    s1[tid] = a; s2[tid] = q;
    __syncthreads();
    for (int s = 128; s > 0; s >>= 1) {
        if (tid < s) { s1[tid] += s1[tid + s]; s2[tid] += s2[tid + s]; }
        __syncthreads();
    }
    const float mean = s1[0] / (float)D;
    const float var  = s2[0] / (float)D - mean * mean;
    const float rstd = rsqrtf(var + 1e-5f);
    for (int d = tid; d < D; d += 256) {
        float v = xr[d] + (rr ? rr[d] : 0.0f);
        out[(long)row * D + d] = (v - mean) * rstd * g[d] + b[d];
    }
}

// ---------------------------------------------------------------------------
// intv self-attention (16 queries x 16 keys per (b,h)). One block per (b,h).
// ---------------------------------------------------------------------------
__global__ __launch_bounds__(256)
void attn_small_kernel(const float* __restrict__ Q, const float* __restrict__ Kt,
                       const float* __restrict__ V, float* __restrict__ outp)
{
    __shared__ float sQ[NUM_SLOTS * HEAD_DIM];
    __shared__ float sK[NUM_SLOTS * HEAD_DIM];
    __shared__ float sV[NUM_SLOTS * HEAD_DIM];
    __shared__ float sP[NUM_SLOTS * NUM_SLOTS];

    const int b = blockIdx.x >> 4;
    const int h = blockIdx.x & 15;
    const int tid = threadIdx.x;

    for (int e = tid; e < NUM_SLOTS * HEAD_DIM; e += 256) {
        int s = e >> 7, d = e & 127;
        long off = (long)(b * NUM_SLOTS + s) * D_MODEL + h * HEAD_DIM + d;
        sQ[e] = Q[off]; sK[e] = Kt[off]; sV[e] = V[off];
    }
    __syncthreads();

    {   // 256 threads -> one score each
        int i = tid >> 4, j = tid & 15;
        float s = 0.0f;
        for (int d = 0; d < HEAD_DIM; ++d) s += sQ[i * HEAD_DIM + d] * sK[j * HEAD_DIM + d];
        sP[i * 16 + j] = s * 0.08838834764831845f;   // 1/sqrt(128)
    }
    __syncthreads();

    if (tid < NUM_SLOTS) {   // softmax per query row
        int i = tid;
        float mx = sP[i * 16];
        for (int j = 1; j < 16; ++j) mx = fmaxf(mx, sP[i * 16 + j]);
        float sum = 0.0f;
        for (int j = 0; j < 16; ++j) { float e = expf(sP[i * 16 + j] - mx); sP[i * 16 + j] = e; sum += e; }
        float inv = 1.0f / sum;
        for (int j = 0; j < 16; ++j) sP[i * 16 + j] *= inv;
    }
    __syncthreads();

    for (int e = tid; e < NUM_SLOTS * HEAD_DIM; e += 256) {
        int i = e >> 7, d = e & 127;
        float acc = 0.0f;
        for (int j = 0; j < 16; ++j) acc += sP[i * 16 + j] * sV[j * HEAD_DIM + d];
        outp[(long)(b * NUM_SLOTS + i) * D_MODEL + h * HEAD_DIM + d] = acc;
    }
}

// ---------------------------------------------------------------------------
// intervention[b,j,d] = sum_i edge[b,i,j] * messages[b,i,d]
// ---------------------------------------------------------------------------
__global__ __launch_bounds__(256)
void intervention_kernel(const float* __restrict__ edge, const float* __restrict__ msg,
                         float* __restrict__ out)
{
    int idx = blockIdx.x * blockDim.x + threadIdx.x;          // B*16*2048
    if (idx >= BATCH * NUM_SLOTS * D_MODEL) return;
    int b = idx / (NUM_SLOTS * D_MODEL);
    int rem = idx - b * NUM_SLOTS * D_MODEL;
    int j = rem / D_MODEL;
    int d = rem - j * D_MODEL;
    float acc = 0.0f;
    for (int i = 0; i < NUM_SLOTS; ++i)
        acc += edge[b * 256 + i * 16 + j] * msg[(long)(b * NUM_SLOTS + i) * D_MODEL + d];
    out[idx] = acc;
}

// ---------------------------------------------------------------------------
// bcast attention: scores+softmax. One thread per (b,h,q). Qp shared over b.
// ---------------------------------------------------------------------------
__global__ __launch_bounds__(256)
void bcast_scores_kernel(const float* __restrict__ Qp, const float* __restrict__ Kb,
                         float* __restrict__ attn)
{
    int idx = blockIdx.x * blockDim.x + threadIdx.x;          // B*H*512
    if (idx >= BATCH * NUM_HEADS * SEQ_LEN) return;
    int b = idx / (NUM_HEADS * SEQ_LEN);
    int rem = idx - b * NUM_HEADS * SEQ_LEN;
    int h = rem / SEQ_LEN;
    int q = rem - h * SEQ_LEN;

    float s[NUM_SLOTS];
    float mx = -1e30f;
    for (int j = 0; j < NUM_SLOTS; ++j) {
        const float* qp = Qp + (long)q * D_MODEL + h * HEAD_DIM;
        const float* kp = Kb + (long)(b * NUM_SLOTS + j) * D_MODEL + h * HEAD_DIM;
        float acc = 0.0f;
        for (int d = 0; d < HEAD_DIM; ++d) acc += qp[d] * kp[d];
        acc *= 0.08838834764831845f;
        s[j] = acc;
        mx = fmaxf(mx, acc);
    }
    float sum = 0.0f;
    for (int j = 0; j < NUM_SLOTS; ++j) { s[j] = expf(s[j] - mx); sum += s[j]; }
    float inv = 1.0f / sum;
    long base = ((long)(b * NUM_HEADS + h) * SEQ_LEN + q) * NUM_SLOTS;
    for (int j = 0; j < NUM_SLOTS; ++j) attn[base + j] = s[j] * inv;
}

// out[b,q,h*128+d] = sum_j attn[b,h,q,j] * Vb[b,j,h*128+d]
__global__ __launch_bounds__(256)
void bcast_out_kernel(const float* __restrict__ attn, const float* __restrict__ Vb,
                      float* __restrict__ outp)
{
    long idx = (long)blockIdx.x * blockDim.x + threadIdx.x;   // B*512*2048
    if (idx >= (long)BATCH * SEQ_LEN * D_MODEL) return;
    int b = (int)(idx / ((long)SEQ_LEN * D_MODEL));
    long rem = idx - (long)b * SEQ_LEN * D_MODEL;
    int q = (int)(rem / D_MODEL);
    int c = (int)(rem - (long)q * D_MODEL);
    int h = c >> 7;
    long abase = ((long)(b * NUM_HEADS + h) * SEQ_LEN + q) * NUM_SLOTS;
    float acc = 0.0f;
    for (int j = 0; j < NUM_SLOTS; ++j)
        acc += attn[abase + j] * Vb[(long)(b * NUM_SLOTS + j) * D_MODEL + c];
    outp[idx] = acc;
}

// ---------------------------------------------------------------------------
// Host launcher
// ---------------------------------------------------------------------------
extern "C" void kernel_launch(void* const* d_in, const int* in_sizes, int n_in,
                              void* d_out, int out_size, void* d_ws, size_t ws_size,
                              hipStream_t stream) {
    (void)in_sizes; (void)n_in; (void)out_size; (void)ws_size;

    const float* slots   = (const float*)d_in[0];   // (8,16,2048)
    const float* content = (const float*)d_in[1];   // (8,2048)
    const float* edge    = (const float*)d_in[2];   // (8,16,16)
    const float* dec     = (const float*)d_in[3];   // (8,512,2048)
    auto P = [&](int i) -> const float* { return (const float*)d_in[4 + i]; };

    enum { BIND = 0, INTV = 10, BCAST = 20 };
    const int W_Q = 0, B_Q = 1, W_K = 2, B_K = 3, W_V = 4, B_V = 5, W_O = 6, B_O = 7, LN_G = 8, LN_B = 9;
    const int AD1_W = 30, AD1_B = 31, AD_LN_G = 32, AD_LN_B = 33,
              AD2_W = 34, AD2_B = 35, GATE_W = 36, GATE_B = 37, POS_Q = 38;

    // ---- workspace bump allocator
    char* base = (char*)d_ws;
    size_t off = 0;
    auto alloc = [&](size_t bytes) -> void* {
        void* p = base + off;
        off = (off + bytes + 255) & ~(size_t)255;
        return p;
    };
    const size_t W22 = (size_t)D_MODEL * D_MODEL;        // 2048x2048
    const size_t W42 = (size_t)2 * D_MODEL * D_MODEL;    // 4096x2048

    _Float16* w_bind_v = (_Float16*)alloc(W22 * 2);
    _Float16* w_bind_o = (_Float16*)alloc(W22 * 2);
    _Float16* w_intv_q = (_Float16*)alloc(W22 * 2);
    _Float16* w_intv_k = (_Float16*)alloc(W22 * 2);
    _Float16* w_intv_v = (_Float16*)alloc(W22 * 2);
    _Float16* w_intv_o = (_Float16*)alloc(W22 * 2);
    _Float16* w_bc_q   = (_Float16*)alloc(W22 * 2);
    _Float16* w_bc_k   = (_Float16*)alloc(W22 * 2);
    _Float16* w_bc_v   = (_Float16*)alloc(W22 * 2);
    _Float16* w_bc_o   = (_Float16*)alloc(W22 * 2);
    _Float16* w_ad1    = (_Float16*)alloc(W42 * 2);
    _Float16* w_ad2    = (_Float16*)alloc(W22 * 2);
    _Float16* w_gate   = (_Float16*)alloc(W42 * 2);

    const size_t SLOT_F = (size_t)BATCH * NUM_SLOTS * D_MODEL;   // 262144
    const size_t BIG_F  = (size_t)BATCH * SEQ_LEN * D_MODEL;     // 8388608

    float* tmpV     = (float*)alloc((size_t)BATCH * D_MODEL * 4);
    float* bindout  = (float*)alloc((size_t)BATCH * D_MODEL * 4);
    float* bound    = (float*)alloc(SLOT_F * 4);
    float* qi       = (float*)alloc(SLOT_F * 4);
    float* ki       = (float*)alloc(SLOT_F * 4);
    float* vi       = (float*)alloc(SLOT_F * 4);
    float* aouti    = (float*)alloc(SLOT_F * 4);
    float* msg      = (float*)alloc(SLOT_F * 4);
    float* intv     = (float*)alloc(SLOT_F * 4);
    float* bound2   = (float*)alloc(SLOT_F * 4);
    float* Qp       = (float*)alloc((size_t)SEQ_LEN * D_MODEL * 4);
    float* Kb       = (float*)alloc(SLOT_F * 4);
    float* Vb       = (float*)alloc(SLOT_F * 4);
    float* attn_b   = (float*)alloc((size_t)BATCH * NUM_HEADS * SEQ_LEN * NUM_SLOTS * 4);
    float* aout_b   = (float*)alloc(BIG_F * 4);
    float* bcast    = (float*)alloc(BIG_F * 4);
    float* gatebuf  = (float*)alloc(BIG_F * 4);
    float* hbuf     = (float*)alloc(BIG_F * 4);

    // ---- 1. convert + transpose weights to f16 (dst is N x K)
    auto cvt = [&](const float* s, _Float16* d, int K, int N) {
        dim3 grid(N / 32, K / 32);
        cvt_transpose_f16<<<grid, 256, 0, stream>>>(s, d, K, N);
    };
    cvt(P(BIND + W_V), w_bind_v, D_MODEL, D_MODEL);
    cvt(P(BIND + W_O), w_bind_o, D_MODEL, D_MODEL);
    cvt(P(INTV + W_Q), w_intv_q, D_MODEL, D_MODEL);
    cvt(P(INTV + W_K), w_intv_k, D_MODEL, D_MODEL);
    cvt(P(INTV + W_V), w_intv_v, D_MODEL, D_MODEL);
    cvt(P(INTV + W_O), w_intv_o, D_MODEL, D_MODEL);
    cvt(P(BCAST + W_Q), w_bc_q, D_MODEL, D_MODEL);
    cvt(P(BCAST + W_K), w_bc_k, D_MODEL, D_MODEL);
    cvt(P(BCAST + W_V), w_bc_v, D_MODEL, D_MODEL);
    cvt(P(BCAST + W_O), w_bc_o, D_MODEL, D_MODEL);
    cvt(P(AD1_W), w_ad1, 2 * D_MODEL, D_MODEL);
    cvt(P(AD2_W), w_ad2, D_MODEL, D_MODEL);
    cvt(P(GATE_W), w_gate, 2 * D_MODEL, D_MODEL);

    auto gemm = [&](const float* A1, int lda1, const float* A2, int lda2, int splitK,
                    const _Float16* WT, const float* bias, float* C,
                    int M, int N, int K, int a_gelu, int act_sig,
                    const float* eg, const float* er) {
        dim3 grid((N + BN - 1) / BN, (M + BM - 1) / BM);
        gemm_wmma_f16<<<grid, 256, 0, stream>>>(A1, lda1, A2, lda2, splitK, WT, bias, C,
                                                M, N, K, a_gelu, act_sig, eg, er);
    };

    // ---- 2. bind MHA (nk=1 => softmax==1 => out = V @ o_W + o_b, broadcast)
    gemm(content, D_MODEL, nullptr, 0, D_MODEL, w_bind_v, P(BIND + B_V), tmpV,
         BATCH, D_MODEL, D_MODEL, 0, 0, nullptr, nullptr);
    gemm(tmpV, D_MODEL, nullptr, 0, D_MODEL, w_bind_o, P(BIND + B_O), bindout,
         BATCH, D_MODEL, D_MODEL, 0, 0, nullptr, nullptr);
    ln_kernel<<<BATCH * NUM_SLOTS, 256, 0, stream>>>(slots, bindout, NUM_SLOTS,
        P(BIND + LN_G), P(BIND + LN_B), bound, D_MODEL);

    // ---- 3. intv self-attention over 16 slots
    const int MS = BATCH * NUM_SLOTS;  // 128
    gemm(bound, D_MODEL, nullptr, 0, D_MODEL, w_intv_q, P(INTV + B_Q), qi,
         MS, D_MODEL, D_MODEL, 0, 0, nullptr, nullptr);
    gemm(bound, D_MODEL, nullptr, 0, D_MODEL, w_intv_k, P(INTV + B_K), ki,
         MS, D_MODEL, D_MODEL, 0, 0, nullptr, nullptr);
    gemm(bound, D_MODEL, nullptr, 0, D_MODEL, w_intv_v, P(INTV + B_V), vi,
         MS, D_MODEL, D_MODEL, 0, 0, nullptr, nullptr);
    attn_small_kernel<<<BATCH * NUM_HEADS, 256, 0, stream>>>(qi, ki, vi, aouti);
    gemm(aouti, D_MODEL, nullptr, 0, D_MODEL, w_intv_o, P(INTV + B_O), msg,
         MS, D_MODEL, D_MODEL, 0, 0, nullptr, nullptr);
    intervention_kernel<<<(BATCH * NUM_SLOTS * D_MODEL + 255) / 256, 256, 0, stream>>>(
        edge, msg, intv);
    ln_kernel<<<BATCH * NUM_SLOTS, 256, 0, stream>>>(bound, intv, 1,
        P(INTV + LN_G), P(INTV + LN_B), bound2, D_MODEL);

    // ---- 4. bcast cross-attention (pos_q is batch-broadcast: project once)
    gemm(P(POS_Q), D_MODEL, nullptr, 0, D_MODEL, w_bc_q, P(BCAST + B_Q), Qp,
         SEQ_LEN, D_MODEL, D_MODEL, 0, 0, nullptr, nullptr);
    gemm(bound2, D_MODEL, nullptr, 0, D_MODEL, w_bc_k, P(BCAST + B_K), Kb,
         MS, D_MODEL, D_MODEL, 0, 0, nullptr, nullptr);
    gemm(bound2, D_MODEL, nullptr, 0, D_MODEL, w_bc_v, P(BCAST + B_V), Vb,
         MS, D_MODEL, D_MODEL, 0, 0, nullptr, nullptr);
    bcast_scores_kernel<<<(BATCH * NUM_HEADS * SEQ_LEN + 255) / 256, 256, 0, stream>>>(
        Qp, Kb, attn_b);
    {
        long n = (long)BATCH * SEQ_LEN * D_MODEL;
        bcast_out_kernel<<<(unsigned)((n + 255) / 256), 256, 0, stream>>>(attn_b, Vb, aout_b);
    }
    const int MB = BATCH * SEQ_LEN;   // 4096
    gemm(aout_b, D_MODEL, nullptr, 0, D_MODEL, w_bc_o, P(BCAST + B_O), bcast,
         MB, D_MODEL, D_MODEL, 0, 0, nullptr, nullptr);
    ln_kernel<<<MB, 256, 0, stream>>>(bcast, nullptr, 1,
        P(BCAST + LN_G), P(BCAST + LN_B), bcast, D_MODEL);   // in-place

    // ---- 5. adapter: combined = [dec | bcast] (virtual concat via split-K)
    gemm(dec, D_MODEL, bcast, D_MODEL, D_MODEL, w_gate, P(GATE_B), gatebuf,
         MB, D_MODEL, 2 * D_MODEL, 0, /*sigmoid*/1, nullptr, nullptr);
    gemm(dec, D_MODEL, bcast, D_MODEL, D_MODEL, w_ad1, P(AD1_B), hbuf,
         MB, D_MODEL, 2 * D_MODEL, 0, 0, nullptr, nullptr);
    ln_kernel<<<MB, 256, 0, stream>>>(hbuf, nullptr, 1,
        P(AD_LN_G), P(AD_LN_B), hbuf, D_MODEL);              // in-place

    // ---- 6. adapted = gelu(h) @ ad2 + b ; fused blend -> d_out
    gemm(hbuf, D_MODEL, nullptr, 0, D_MODEL, w_ad2, P(AD2_B), (float*)d_out,
         MB, D_MODEL, D_MODEL, /*gelu on A*/1, 0, gatebuf, dec);
}
